// Attention_5068061409491
// MI455X (gfx1250) — matrix operand order repaired and checked by reading, not compiled, not run on hardware
//
#include <hip/hip_runtime.h>

// ---------------------------------------------------------------------------
// MI455X / gfx1250 fused attention:
//   Phase 1: Q/K/V = x@W + b  (fp32 -> bf16, v_wmma_f32_16x16x32_bf16)
//            V written TRANSPOSED [B][DV][SEQ] so phase 2 needs no transpose.
//   Phase 2: flash attention, async global->LDS staging (ASYNCcnt),
//            WMMA for QK^T and PV, online softmax.
// ---------------------------------------------------------------------------

typedef __attribute__((ext_vector_type(16))) __bf16 v16bf;
typedef __attribute__((ext_vector_type(8)))  __bf16 bf16x8;
typedef __attribute__((ext_vector_type(8)))  float  v8f;
typedef int v4i __attribute__((__vector_size__(16)));   // matches builtin proto

#define WMMA_BF16(A, B, C) \
  __builtin_amdgcn_wmma_f32_16x16x32_bf16(false, (A), false, (B), (short)0, (C), false, false)

#define AS1 __attribute__((address_space(1)))
#define AS3 __attribute__((address_space(3)))

#if defined(__AMDGCN__) && __has_builtin(__builtin_amdgcn_global_load_async_to_lds_b128)
#define HAS_ASYNC_LDS 1
#else
#define HAS_ASYNC_LDS 0
#endif

static constexpr int BATCH = 8;
static constexpr int SEQ   = 2048;
static constexpr int DIM   = 512;            // D == DK == DV
static constexpr int ROWS  = BATCH * SEQ;    // 16384

// 16-byte global->LDS copy: async (no VGPR bounce, ASYNCcnt) when available.
__device__ __forceinline__ void cp_b128(__bf16* ldst, const __bf16* gsrc) {
#if HAS_ASYNC_LDS
  __builtin_amdgcn_global_load_async_to_lds_b128(
      (AS1 v4i*)gsrc, (AS3 v4i*)ldst, 0, 0);
#else
  *(bf16x8*)ldst = *(const bf16x8*)gsrc;
#endif
}

__device__ __forceinline__ void cp_wait() {
#if HAS_ASYNC_LDS
#if __has_builtin(__builtin_amdgcn_s_wait_asynccnt)
  __builtin_amdgcn_s_wait_asynccnt(0);
#else
  asm volatile("s_wait_asynccnt 0x0" ::: "memory");
#endif
#endif
}

// ---------------------------------------------------------------------------
// Phase 1: projection GEMM.  grid = (Ntiles=4, Mtiles=128, proj=3), block=256.
// Block tile 128x128; 8 waves, each 32x64 (2x4 WMMA accumulators).
// ---------------------------------------------------------------------------
__global__ __launch_bounds__(256)
void qkv_gemm(const float* __restrict__ x,
              const float* __restrict__ Wq, const float* __restrict__ bq,
              const float* __restrict__ Wk, const float* __restrict__ bk,
              const float* __restrict__ Wv, const float* __restrict__ bv,
              __bf16* __restrict__ ws)
{
  const int proj = blockIdx.z;
  const float* W    = (proj == 0) ? Wq : (proj == 1) ? Wk : Wv;
  const float* bias = (proj == 0) ? bq : (proj == 1) ? bk : bv;

  const int m0  = blockIdx.y * 128;
  const int n0c = blockIdx.x * 128;

  __shared__ __bf16 xs[128 * 32];   // [row][k]  row-major, stride 32
  __shared__ __bf16 wt[128 * 32];   // [n][k]    transposed W tile, stride 32

  const int t    = threadIdx.x;
  const int lane = t & 31;
  const int wid  = t >> 5;
  const int ln   = lane & 15;
  const int hk   = lane >> 4;
  const int wm   = (wid & 3) * 32;
  const int wn   = (wid >> 2) * 64;

  v8f acc[2][4];
#pragma unroll
  for (int mf = 0; mf < 2; ++mf)
#pragma unroll
    for (int nf = 0; nf < 4; ++nf) acc[mf][nf] = {};

  for (int kc = 0; kc < DIM; kc += 32) {
    // stage x tile (128x32 fp32 -> bf16), packed 16B LDS stores
    {
      const int row = t >> 1;
      const int c0  = (t & 1) * 16;
      const float* src = x + (size_t)(m0 + row) * DIM + kc + c0;
#pragma unroll
      for (int h = 0; h < 2; ++h) {
        bf16x8 v;
#pragma unroll
        for (int j = 0; j < 8; ++j) v[j] = (__bf16)src[h * 8 + j];
        *(bf16x8*)&xs[row * 32 + c0 + h * 8] = v;
      }
    }
    // stage W tile transposed (read [k][n] coalesced, write [n][k])
    {
      const int k  = t >> 3;
      const int nb = (t & 7) * 16;
      const float* src = W + (size_t)(kc + k) * DIM + n0c + nb;
#pragma unroll
      for (int j = 0; j < 16; ++j) wt[(nb + j) * 32 + k] = (__bf16)src[j];
    }
    __syncthreads();

    v16bf a[2];
#pragma unroll
    for (int mf = 0; mf < 2; ++mf) {
      const __bf16* p = &xs[(wm + mf * 16 + ln) * 32 + hk * 8];
      bf16x8 lo = *(const bf16x8*)p;
      bf16x8 hi = *(const bf16x8*)(p + 16);
#pragma unroll
      for (int i = 0; i < 8; ++i) { a[mf][i] = lo[i]; a[mf][i + 8] = hi[i]; }
    }
#pragma unroll
    for (int nf = 0; nf < 4; ++nf) {
      const __bf16* p = &wt[(wn + nf * 16 + ln) * 32 + hk * 8];
      bf16x8 lo = *(const bf16x8*)p;
      bf16x8 hi = *(const bf16x8*)(p + 16);
      v16bf b;
#pragma unroll
      for (int i = 0; i < 8; ++i) { b[i] = lo[i]; b[i + 8] = hi[i]; }
#pragma unroll
      for (int mf = 0; mf < 2; ++mf)
        acc[mf][nf] = WMMA_BF16(a[mf], b, acc[mf][nf]);
    }
    __syncthreads();
  }

  if (proj != 2) {
    // Q/K: row-major [ROWS][DIM] bf16
    __bf16* out = ws + (size_t)proj * ((size_t)ROWS * DIM);
#pragma unroll
    for (int nf = 0; nf < 4; ++nf) {
      const int col = n0c + wn + nf * 16 + ln;
      const float bval = bias[col];
#pragma unroll
      for (int mf = 0; mf < 2; ++mf) {
#pragma unroll
        for (int r = 0; r < 8; ++r) {
          const int row = m0 + wm + mf * 16 + r + 8 * hk;   // C layout
          out[(size_t)row * DIM + col] = (__bf16)(acc[mf][nf][r] + bval);
        }
      }
    }
  } else {
    // V: TRANSPOSED [B][DIM][SEQ].  C-layout gives each lane 8 consecutive
    // rows (r + 8*hk), contiguous in the transposed layout -> one packed
    // 16-byte store per accumulator tile.
    const int batch = m0 / SEQ;                 // 128-row tiles never straddle
    const int rbase = (m0 % SEQ) + wm;
    __bf16* outv = ws + 2ull * ROWS * DIM + (size_t)batch * DIM * SEQ;
#pragma unroll
    for (int nf = 0; nf < 4; ++nf) {
      const int col = n0c + wn + nf * 16 + ln;
      const float bval = bias[col];
#pragma unroll
      for (int mf = 0; mf < 2; ++mf) {
        bf16x8 w;
#pragma unroll
        for (int r = 0; r < 8; ++r) w[r] = (__bf16)(acc[mf][nf][r] + bval);
        *(bf16x8*)(outv + (size_t)col * SEQ + rbase + mf * 16 + 8 * hk) = w;
      }
    }
  }
}

// ---------------------------------------------------------------------------
// Phase 2: flash attention.  grid = (64 Mtiles, 8 batches), block=256.
// Block = 32 query rows; waves = 2 M-subtiles x 4 DV-slices(128).
// K tile (32x512, contiguous) async-staged into LDS; V tile read from the
// transposed workspace (contiguous 64B rows) async-staged into the SAME
// buffer after the S phase.  All WMMA fragment reads are ds_load_b128.
// ---------------------------------------------------------------------------
__global__ __launch_bounds__(256)
void attn(const __bf16* __restrict__ ws, float* __restrict__ out)
{
  const int b  = blockIdx.y;
  const int mt = blockIdx.x;

  const __bf16* Q  = ws;
  const __bf16* K  = ws + (size_t)ROWS * DIM;
  const __bf16* Vt = ws + 2ull * ROWS * DIM + (size_t)b * DIM * SEQ; // [DIM][SEQ]

  __shared__ __bf16 kv[32 * 512];      // 32 KB: K row-major, then Vt tile [512][32]
  __shared__ __bf16 plds[8 * 16 * 32]; // 8 KB: per-wave P bounce (C -> A layout)

  const int t     = threadIdx.x;
  const int lane  = t & 31;
  const int wid   = t >> 5;
  const int ln    = lane & 15;
  const int hk    = lane >> 4;
  const int mhalf = wid & 1;
  const int dv0   = (wid >> 1) * 128;

  const int rowbase = b * SEQ + mt * 32;

  // ---- Q fragments for this wave's 16 rows held in registers --------------
  v16bf qf[16];
  {
    const __bf16* qrow = Q + (size_t)(rowbase + mhalf * 16 + ln) * DIM;
#pragma unroll
    for (int kc = 0; kc < 16; ++kc) {
      const __bf16* p = qrow + kc * 32 + hk * 8;
      bf16x8 lo = *(const bf16x8*)p;
      bf16x8 hi = *(const bf16x8*)(p + 16);
#pragma unroll
      for (int i = 0; i < 8; ++i) { qf[kc][i] = lo[i]; qf[kc][i + 8] = hi[i]; }
    }
  }

  v8f acc[8];
#pragma unroll
  for (int nf = 0; nf < 8; ++nf) acc[nf] = {};
  float m[8], l[8];
#pragma unroll
  for (int r = 0; r < 8; ++r) { m[r] = -1.0e30f; l[r] = 0.0f; }

  const float inv_norm = 22.627416997969522f;  // sqrt(DK): post-softmax scale

  for (int kb = 0; kb < SEQ / 32; ++kb) {
    // ---- async stage K tile: 32KB fully contiguous in ws ----
    {
      const __bf16* src = K + (size_t)(b * SEQ + kb * 32) * DIM + t * 64;
      __bf16* dst = kv + t * 64;
#pragma unroll
      for (int j = 0; j < 8; ++j) cp_b128(dst + j * 8, src + j * 8);
    }
    cp_wait();
    __syncthreads();

    // ---- S = Q @ K^T (16 x 32): B[k][n] = K[key n][k] -> contiguous b128 --
    v8f s[2];
    s[0] = {}; s[1] = {};
#pragma unroll
    for (int kc = 0; kc < 16; ++kc) {
#pragma unroll
      for (int nt = 0; nt < 2; ++nt) {
        const __bf16* p = kv + (size_t)(nt * 16 + ln) * 512 + kc * 32 + hk * 8;
        bf16x8 lo = *(const bf16x8*)p;
        bf16x8 hi = *(const bf16x8*)(p + 16);
        v16bf bb;
#pragma unroll
        for (int i = 0; i < 8; ++i) { bb[i] = lo[i]; bb[i + 8] = hi[i]; }
        s[nt] = WMMA_BF16(qf[kc], bb, s[nt]);
      }
    }

    // ---- online softmax (rows live in half-waves; xor-shuffle reductions) --
#pragma unroll
    for (int r = 0; r < 8; ++r) {
      float mx = fmaxf(s[0][r], s[1][r]);
      mx = fmaxf(mx, __shfl_xor(mx, 1));
      mx = fmaxf(mx, __shfl_xor(mx, 2));
      mx = fmaxf(mx, __shfl_xor(mx, 4));
      mx = fmaxf(mx, __shfl_xor(mx, 8));
      const float mn = fmaxf(m[r], mx);
      const float fac = __expf(m[r] - mn);
      m[r] = mn;
      const float p0 = __expf(s[0][r] - mn);
      const float p1 = __expf(s[1][r] - mn);
      plds[wid * 512 + (r + 8 * hk) * 32 + ln]      = (__bf16)p0;
      plds[wid * 512 + (r + 8 * hk) * 32 + 16 + ln] = (__bf16)p1;
      float sum = p0 + p1;
      sum += __shfl_xor(sum, 1);
      sum += __shfl_xor(sum, 2);
      sum += __shfl_xor(sum, 4);
      sum += __shfl_xor(sum, 8);
      l[r] = l[r] * fac + sum;
#pragma unroll
      for (int nf = 0; nf < 8; ++nf) acc[nf][r] *= fac;
    }

    // ---- re-read P in A-fragment layout (same-wave DS ops are in-order) ----
    v16bf pf;
    {
      const __bf16* p = plds + wid * 512 + ln * 32 + hk * 8;
      bf16x8 lo = *(const bf16x8*)p;
      bf16x8 hi = *(const bf16x8*)(p + 16);
#pragma unroll
      for (int i = 0; i < 8; ++i) { pf[i] = lo[i]; pf[i + 8] = hi[i]; }
    }

    __syncthreads();  // everyone done reading K tile

    // ---- async stage V tile from transposed ws: kv = [col 0..511][key 0..31]
    {
      const int c = t * 2;                      // 2 columns per thread
#pragma unroll
      for (int cc = 0; cc < 2; ++cc) {
        const __bf16* src = Vt + (size_t)(c + cc) * SEQ + kb * 32;
        __bf16* dst = kv + (c + cc) * 32;
#pragma unroll
        for (int j = 0; j < 4; ++j) cp_b128(dst + j * 8, src + j * 8);
      }
    }
    // prefetch next K tile (global_prefetch_b8)
    if (kb + 1 < SEQ / 32) {
      __builtin_prefetch(K + (size_t)(b * SEQ + (kb + 1) * 32) * DIM + t * 64, 0, 0);
    }
    cp_wait();
    __syncthreads();

    // ---- O += P @ V_slice : B[k][n] = Vt[col n][key k] -> contiguous b128 --
#pragma unroll
    for (int nf = 0; nf < 8; ++nf) {
      const __bf16* p = kv + (size_t)(dv0 + nf * 16 + ln) * 32 + hk * 8;
      bf16x8 lo = *(const bf16x8*)p;
      bf16x8 hi = *(const bf16x8*)(p + 16);
      v16bf bb;
#pragma unroll
      for (int i = 0; i < 8; ++i) { bb[i] = lo[i]; bb[i + 8] = hi[i]; }
      acc[nf] = WMMA_BF16(pf, bb, acc[nf]);
    }
    __syncthreads();  // done reading V before next K staging
  }

  // ---- epilogue: out = acc * sqrt(DK) / l ----
#pragma unroll
  for (int r = 0; r < 8; ++r) {
    const float scale = inv_norm / l[r];
    const int row = rowbase + mhalf * 16 + r + 8 * hk;  // C layout
#pragma unroll
    for (int nf = 0; nf < 8; ++nf)
      out[(size_t)row * DIM + dv0 + nf * 16 + ln] = acc[nf][r] * scale;
  }
}

// ---------------------------------------------------------------------------
extern "C" void kernel_launch(void* const* d_in, const int* in_sizes, int n_in,
                              void* d_out, int out_size, void* d_ws, size_t ws_size,
                              hipStream_t stream) {
  const float* x  = (const float*)d_in[0];
  const float* Wq = (const float*)d_in[1];
  const float* bq = (const float*)d_in[2];
  const float* Wk = (const float*)d_in[3];
  const float* bk = (const float*)d_in[4];
  const float* Wv = (const float*)d_in[5];
  const float* bv = (const float*)d_in[6];
  float*  out = (float*)d_out;
  __bf16* ws  = (__bf16*)d_ws;   // needs 3 * 16384 * 512 * 2 = 48 MiB

  qkv_gemm<<<dim3(4, 128, 3), 256, 0, stream>>>(x, Wq, bq, Wk, bk, Wv, bv, ws);
  attn<<<dim3(64, 8), 256, 0, stream>>>(ws, out);
}
